// StockBlockLayer_52553219833972
// MI455X (gfx1250) — compile-verified, matrix operand order checked by reading, not run on hardware
//
#include <hip/hip_runtime.h>

// ---------------- WMMA types / helper (gfx1250, wave32) ----------------
typedef __attribute__((ext_vector_type(16))) __bf16 v16bf;
typedef __attribute__((ext_vector_type(8)))  float  v8f;

__device__ __forceinline__ v8f wmma_bf16(v16bf a, v16bf b, v8f c) {
  // D = A(16x32 bf16) x B(32x16 bf16) + C(16x16 f32)
  return __builtin_amdgcn_wmma_f32_16x16x32_bf16(false, a, false, b, (short)0, c,
                                                 false, false);
}

// problem constants
constexpr int B_ = 32, N_ = 2048, T_ = 12, MT_ = 60;
constexpr int ROWS = B_ * N_;                // 65536 data rows

// ---------------- prep: convert mul_l to bf16 ----------------
__global__ void k_cvt_mul(const float* __restrict__ mul, __bf16* __restrict__ mulbf,
                          int nElem) {
  int i = blockIdx.x * 256 + threadIdx.x;
  if (i < nElem) mulbf[i] = (__bf16)mul[i];
}

// ---------------- prep: x (B,N,12) -> xT (B,16,N) bf16, zero pad t>=12 ----
__global__ void k_build_xT(const float* __restrict__ x, __bf16* __restrict__ xT) {
  int i = blockIdx.x * 256 + threadIdx.x;       // B*16*N = 1048576
  if (i >= B_ * 16 * N_) return;
  int j = i & (N_ - 1);
  int t = (i >> 11) & 15;
  int b = i >> 15;
  float v = (t < T_) ? x[((size_t)b * N_ + j) * T_ + t] : 0.f;
  xT[i] = (__bf16)v;
}

// ------- prep: (rows, fin) f32 -> (rows, finPad) bf16, zero-padded --------
__global__ void k_cvt_pad(const float* __restrict__ src, __bf16* __restrict__ dst,
                          int rows, int fin, int finPad) {
  int i = blockIdx.x * 256 + threadIdx.x;
  if (i >= rows * finPad) return;
  int r = i / finPad, c = i - r * finPad;
  dst[i] = (c < fin) ? (__bf16)src[(size_t)r * fin + c] : (__bf16)0.f;
}

// ------- prep: weight (4,60,60)[k,p,m] -> bf16 (4,64,64)[k,m,p] padded ----
__global__ void k_cvt_w4(const float* __restrict__ w, __bf16* __restrict__ wT) {
  int i = blockIdx.x * 256 + threadIdx.x;       // 4*64*64 = 16384
  if (i >= 4 * 64 * 64) return;
  int p = i & 63, m = (i >> 6) & 63, k = i >> 12;
  wT[i] = (p < MT_ && m < MT_) ? (__bf16)w[(k * MT_ + p) * MT_ + m] : (__bf16)0.f;
}

// ---------------- gfted GEMM + length-4 FFT (fused, 4-batch blocked) -------
// block = 128 threads = 4 waves (one per k). grid = (B/4) * (N/16) = 1024
// Each wave reuses its A fragment across 4 batch elements (4 accumulators).
__global__ void k_gfted_fft(const __bf16* __restrict__ mulbf,
                            const __bf16* __restrict__ xT,
                            __bf16* __restrict__ rin, __bf16* __restrict__ iin) {
  __shared__ float lds[4 * 4 * 256];           // [k][bb][m*16 + t]
  const int tid  = threadIdx.x;
  const int wave = tid >> 5;                   // k = 0..3
  const int lane = tid & 31;
  const int bg    = blockIdx.x >> 7;           // batch group 0..7
  const int ntile = blockIdx.x & 127;
  const int l16 = lane & 15;
  const int hi  = (lane >> 4) * 8;

  const __bf16* Arow  = mulbf + ((size_t)wave * N_ + ntile * 16 + l16) * N_;
  const __bf16* Brow0 = xT + ((size_t)(bg * 4) * 16 + l16) * N_;  // l16 = t column

  v8f acc0 = {}, acc1 = {}, acc2 = {}, acc3 = {};
  for (int c = 0; c < N_ / 32; ++c) {
    int k0 = c * 32;
    v16bf a;
#pragma unroll
    for (int i = 0; i < 8; ++i) {
      a[i]     = Arow[k0 + hi + i];
      a[i + 8] = Arow[k0 + 16 + hi + i];
    }
#pragma unroll
    for (int bb = 0; bb < 4; ++bb) {
      const __bf16* Br = Brow0 + (size_t)bb * 16 * N_;
      v16bf bf;
#pragma unroll
      for (int i = 0; i < 8; ++i) {
        bf[i]     = Br[k0 + hi + i];
        bf[i + 8] = Br[k0 + 16 + hi + i];
      }
      if (bb == 0)      acc0 = wmma_bf16(a, bf, acc0);
      else if (bb == 1) acc1 = wmma_bf16(a, bf, acc1);
      else if (bb == 2) acc2 = wmma_bf16(a, bf, acc2);
      else              acc3 = wmma_bf16(a, bf, acc3);
    }
  }
  // C/D layout: n = lane%16 (=t), m = r + 8*(lane>=16)
#pragma unroll
  for (int r = 0; r < 8; ++r) {
    int m = r + hi;
    lds[(wave * 4 + 0) * 256 + m * 16 + l16] = acc0[r];
    lds[(wave * 4 + 1) * 256 + m * 16 + l16] = acc1[r];
    lds[(wave * 4 + 2) * 256 + m * 16 + l16] = acc2[r];
    lds[(wave * 4 + 3) * 256 + m * 16 + l16] = acc3[r];
  }
  __syncthreads();

  for (int idx = tid; idx < 4 * 16 * 64; idx += 128) {
    int bb  = idx >> 10;
    int m   = (idx >> 6) & 15;
    int col = idx & 63;
    float rv = 0.f, iv = 0.f;
    if (col < 48) {
      int cch = col / T_, t = col % T_;
      float z0 = lds[(0 * 4 + bb) * 256 + m * 16 + t];
      float z1 = lds[(1 * 4 + bb) * 256 + m * 16 + t];
      float z2 = lds[(2 * 4 + bb) * 256 + m * 16 + t];
      float z3 = lds[(3 * 4 + bb) * 256 + m * 16 + t];
      if (cch == 0)      { rv = z0 + z1 + z2 + z3; }
      else if (cch == 1) { rv = z0 - z2;            iv = z3 - z1; }
      else if (cch == 2) { rv = z0 - z1 + z2 - z3; }
      else               { rv = z0 - z2;            iv = z1 - z3; }
    }
    size_t outRow = (size_t)(bg * 4 + bb) * N_ + ntile * 16 + m;
    rin[outRow * 64 + col] = (__bf16)rv;
    iin[outRow * 64 + col] = (__bf16)iv;
  }
}

// ---------------- generic GLU layer: out = (in@Wl^T+bl) * sigmoid(in@Wr^T+br)
// in: bf16 (ROWS, inStride) zero-padded; Wl/Wr: bf16 (240, nch*32) pre-padded
// out: bf16 (ROWS, 256). one wave per 16x16 tile; 15 x 4096 = 61440 waves.
__global__ void k_glu(const __bf16* __restrict__ in, int inStride, int nch,
                      const __bf16* __restrict__ Wl, const float* __restrict__ bl,
                      const __bf16* __restrict__ Wr, const float* __restrict__ br,
                      __bf16* __restrict__ out) {
  const int tid = threadIdx.x, lane = tid & 31, wave = tid >> 5;
  int gw = blockIdx.x * 8 + wave;
  int ntile = gw % 15;
  int rowtile = gw / 15;
  const int l16 = lane & 15, hi = (lane >> 4) * 8;
  int mbase = rowtile * 16;
  int n = ntile * 16 + l16;                 // output feature (<240)
  const int wstride = nch * 32;
  const __bf16* Arow  = in + (size_t)(mbase + l16) * inStride;
  const __bf16* wlrow = Wl + (size_t)n * wstride;
  const __bf16* wrrow = Wr + (size_t)n * wstride;

  v8f accl = {}, accr = {};
  for (int c = 0; c < nch; ++c) {
    int k0 = c * 32;
    v16bf a, bL, bR;
#pragma unroll
    for (int i = 0; i < 8; ++i) {
      int ka = k0 + hi + i, kb = k0 + 16 + hi + i;
      a[i]      = Arow[ka];
      a[i + 8]  = Arow[kb];
      bL[i]     = wlrow[ka];
      bL[i + 8] = wlrow[kb];
      bR[i]     = wrrow[ka];
      bR[i + 8] = wrrow[kb];
    }
    accl = wmma_bf16(a, bL, accl);
    accr = wmma_bf16(a, bR, accr);
  }
  float blv = bl[n], brv = br[n];
#pragma unroll
  for (int r = 0; r < 8; ++r) {
    int m = mbase + r + hi;
    float lv = accl[r] + blv;
    float rv = accr[r] + brv;
    float v = lv * (1.f / (1.f + __expf(-rv)));
    out[(size_t)m * 256 + n] = (__bf16)v;
  }
  if (ntile == 0) {                          // zero pad cols 240..255 once per row-tile
    for (int idx = lane; idx < 256; idx += 32) {
      int m = mbase + (idx >> 4);
      out[(size_t)m * 256 + 240 + (idx & 15)] = (__bf16)0.f;
    }
  }
}

// ---------------- irfft(n=4) combine -> iff (B,4,N,64) bf16, padded ------
__global__ void k_iff(const __bf16* __restrict__ realf, const __bf16* __restrict__ imagf,
                      __bf16* __restrict__ iff) {
  size_t idx = (size_t)blockIdx.x * 256 + threadIdx.x;   // over ROWS*64
  if (idx >= (size_t)ROWS * 64) return;
  int col = idx & 63;
  size_t rown = idx >> 6;
  float y0 = 0.f, y1 = 0.f, y2 = 0.f, y3 = 0.f;
  if (col < MT_) {
    const __bf16* rp = realf + rown * 256;
    const __bf16* ip = imagf + rown * 256;
    float R0 = (float)rp[col], R1 = (float)rp[MT_ + col], R2 = (float)rp[2 * MT_ + col];
    float I1 = (float)ip[MT_ + col];
    y0 = 0.25f * (R0 + 2.f * R1 + R2);
    y1 = 0.25f * (R0 - 2.f * I1 - R2);
    y2 = 0.25f * (R0 - 2.f * R1 + R2);
    y3 = 0.25f * (R0 + 2.f * I1 - R2);
  }
  size_t b = rown >> 11, n = rown & (N_ - 1);
  iff[(((b * 4 + 0) * N_) + n) * 64 + col] = (__bf16)y0;
  iff[(((b * 4 + 1) * N_) + n) * 64 + col] = (__bf16)y1;
  iff[(((b * 4 + 2) * N_) + n) * 64 + col] = (__bf16)y2;
  iff[(((b * 4 + 3) * N_) + n) * 64 + col] = (__bf16)y3;
}

// ---------------- ig = sum_k iff[b,k] @ weight[k]  (K = 4*64 = 256) -------
// wT: bf16 (4,64,64) [k][m][p] pre-transposed/padded. ig: (ROWS,64) f32.
__global__ void k_ig(const __bf16* __restrict__ iff, const __bf16* __restrict__ wT,
                     float* __restrict__ ig) {
  const int tid = threadIdx.x, lane = tid & 31, wave = tid >> 5;
  int gw = blockIdx.x * 8 + wave;
  int ntile = gw & 3;
  int rowtile = gw >> 2;
  const int l16 = lane & 15, hi = (lane >> 4) * 8;
  int row0 = rowtile * 16;
  int bb = row0 >> 11, n0 = row0 & (N_ - 1);
  int m = ntile * 16 + l16;

  v8f acc = {};
  for (int c = 0; c < 8; ++c) {
    int kk = c >> 1;
    int p0 = (c & 1) * 32;
    const __bf16* Arow = iff + (((size_t)(bb * 4 + kk) * N_) + n0 + l16) * 64 + p0;
    const __bf16* Brow = wT + ((size_t)kk * 64 + m) * 64 + p0;
    v16bf a, bf;
#pragma unroll
    for (int i = 0; i < 8; ++i) {
      int pa = hi + i, pb = 16 + hi + i;
      a[i]      = Arow[pa];
      a[i + 8]  = Arow[pb];
      bf[i]     = Brow[pa];
      bf[i + 8] = Brow[pb];
    }
    acc = wmma_bf16(a, bf, acc);
  }
#pragma unroll
  for (int r = 0; r < 8; ++r) {
    int grow = row0 + r + hi;
    ig[(size_t)grow * 64 + ntile * 16 + l16] = acc[r];
  }
}

// ---------------- heads: forecast / backcast --------------------------------
__global__ void k_final(const float* __restrict__ ig, const float* __restrict__ x,
                        const float* __restrict__ fw_w, const float* __restrict__ fw_b,
                        const float* __restrict__ fr_w, const float* __restrict__ fr_b,
                        const float* __restrict__ bc_w, const float* __restrict__ bc_b,
                        const float* __restrict__ bs_w, const float* __restrict__ bs_b,
                        float* __restrict__ outF, float* __restrict__ outB) {
  __shared__ float sfw[3600], sfwb[60], sfr[720], sfrb[12],
                   sbc[720], sbcb[12], sbs[144], sbsb[12];
  int tid = threadIdx.x;
  for (int i = tid; i < 3600; i += 256) sfw[i] = fw_w[i];
  for (int i = tid; i < 720; i += 256) { sfr[i] = fr_w[i]; sbc[i] = bc_w[i]; }
  for (int i = tid; i < 144; i += 256) sbs[i] = bs_w[i];
  if (tid < 60) sfwb[tid] = fw_b[tid];
  if (tid < 12) { sfrb[tid] = fr_b[tid]; sbcb[tid] = bc_b[tid]; sbsb[tid] = bs_b[tid]; }
  __syncthreads();

  int row = blockIdx.x * 256 + tid;          // 0..ROWS-1
  float igr[60];
#pragma unroll
  for (int p = 0; p < 60; ++p) igr[p] = ig[(size_t)row * 64 + p];
  float fs[60];
  for (int mm = 0; mm < 60; ++mm) {
    float s = sfwb[mm];
    for (int p = 0; p < 60; ++p) s += igr[p] * sfw[mm * 60 + p];
    fs[mm] = 1.f / (1.f + __expf(-s));
  }
  float xr[12];
#pragma unroll
  for (int t = 0; t < 12; ++t) xr[t] = x[(size_t)row * 12 + t];
  for (int t = 0; t < 12; ++t) {
    float f = sfrb[t], bcv = sbcb[t], bsv = sbsb[t];
    for (int mm = 0; mm < 60; ++mm) {
      f   += fs[mm]  * sfr[t * 60 + mm];
      bcv += igr[mm] * sbc[t * 60 + mm];
    }
    for (int s2 = 0; s2 < 12; ++s2) bsv += xr[s2] * sbs[t * 12 + s2];
    outF[(size_t)row * 12 + t] = f;
    outB[(size_t)row * 12 + t] = 1.f / (1.f + __expf(-(bcv - bsv)));
  }
}

// ---------------- host launcher ----------------
extern "C" void kernel_launch(void* const* d_in, const int* in_sizes, int n_in,
                              void* d_out, int out_size, void* d_ws, size_t ws_size,
                              hipStream_t stream) {
  const float* x     = (const float*)d_in[0];
  const float* mul_l = (const float*)d_in[1];
  const float* weight= (const float*)d_in[2];
  // glu params: 6 groups of (Wl, bl, Wr, br) at indices 3..26
  const float* fw_w = (const float*)d_in[27];
  const float* fw_b = (const float*)d_in[28];
  const float* fr_w = (const float*)d_in[29];
  const float* fr_b = (const float*)d_in[30];
  const float* bc_w = (const float*)d_in[31];
  const float* bc_b = (const float*)d_in[32];
  const float* bs_w = (const float*)d_in[33];
  const float* bs_b = (const float*)d_in[34];

  // workspace layout
  char* p = (char*)d_ws;
  __bf16* mulbf = (__bf16*)p;               p += (size_t)4 * N_ * N_ * 2;       // 32 MB
  __bf16* xT    = (__bf16*)p;               p += (size_t)B_ * 16 * N_ * 2;      // 2 MB
  __bf16* rin   = (__bf16*)p;               p += (size_t)ROWS * 64 * 2;         // 8 MB
  __bf16* iin   = (__bf16*)p;               p += (size_t)ROWS * 64 * 2;         // 8 MB
  __bf16* bufA  = (__bf16*)p;               p += (size_t)ROWS * 256 * 2;        // 32 MB
  __bf16* bufB  = (__bf16*)p;               p += (size_t)ROWS * 256 * 2;        // 32 MB
  __bf16* bufC  = (__bf16*)p;               p += (size_t)ROWS * 256 * 2;        // 32 MB
  __bf16* iff   = (__bf16*)p;               p += (size_t)B_ * 4 * N_ * 64 * 2;  // 32 MB
  float*  igbuf = (float*)p;                p += (size_t)ROWS * 64 * 4;         // 16 MB
  __bf16* wT4   = (__bf16*)p;               p += (size_t)4 * 64 * 64 * 2;       // 32 KB
  // per-layer pre-converted GLU weights (bf16, padded)
  __bf16* wglu[12];                          // [layer gi][0]=Wl, [1]=Wr
  for (int gi = 0; gi < 6; ++gi) {
    int finPad = (gi < 2) ? 64 : 256;
    wglu[2 * gi + 0] = (__bf16*)p;          p += (size_t)240 * finPad * 2;
    wglu[2 * gi + 1] = (__bf16*)p;          p += (size_t)240 * finPad * 2;
  }

  float* outF = (float*)d_out;
  float* outB = outF + (size_t)ROWS * T_;

  // 1) precision conversion / staging
  k_cvt_mul<<<(4 * N_ * N_ + 255) / 256, 256, 0, stream>>>(mul_l, mulbf, 4 * N_ * N_);
  k_build_xT<<<(B_ * 16 * N_ + 255) / 256, 256, 0, stream>>>(x, xT);
  for (int gi = 0; gi < 6; ++gi) {
    int fin = (gi < 2) ? 48 : 240;
    int finPad = (gi < 2) ? 64 : 256;
    int ne = 240 * finPad;
    k_cvt_pad<<<(ne + 255) / 256, 256, 0, stream>>>(
        (const float*)d_in[3 + 4 * gi + 0], wglu[2 * gi + 0], 240, fin, finPad);
    k_cvt_pad<<<(ne + 255) / 256, 256, 0, stream>>>(
        (const float*)d_in[3 + 4 * gi + 2], wglu[2 * gi + 1], 240, fin, finPad);
  }
  k_cvt_w4<<<(4 * 64 * 64 + 255) / 256, 256, 0, stream>>>(weight, wT4);

  // 2) gfted GEMM + FFT (fused, 4-batch blocked)
  k_gfted_fft<<<(B_ / 4) * (N_ / 16), 128, 0, stream>>>(mulbf, xT, rin, iin);

  // 3) GLU chains (waves = 4096 row-tiles * 15 n-tiles; 8 waves/block)
  const int gluBlocks = (4096 * 15) / 8;
  auto glu = [&](const __bf16* in, int inStride, int nch, int gi, __bf16* out) {
    const float* bl = (const float*)d_in[3 + 4 * gi + 1];
    const float* br = (const float*)d_in[3 + 4 * gi + 3];
    k_glu<<<gluBlocks, 256, 0, stream>>>(in, inStride, nch,
                                         wglu[2 * gi + 0], bl,
                                         wglu[2 * gi + 1], br, out);
  };
  // real stream: rin -> A -> B -> A
  glu(rin, 64, 2, 0, bufA);
  glu(bufA, 256, 8, 2, bufB);
  glu(bufB, 256, 8, 4, bufA);
  // imag stream: iin -> B -> C -> B
  glu(iin, 64, 2, 1, bufB);
  glu(bufB, 256, 8, 3, bufC);
  glu(bufC, 256, 8, 5, bufB);

  // 4) irfft combine
  k_iff<<<(ROWS * 64 + 255) / 256, 256, 0, stream>>>(bufA, bufB, iff);

  // 5) ig GEMM (waves = 4096 row-tiles * 4 m-tiles; 8 waves/block)
  k_ig<<<(4096 * 4) / 8, 256, 0, stream>>>(iff, wT4, igbuf);

  // 6) heads
  k_final<<<ROWS / 256, 256, 0, stream>>>(igbuf, x, fw_w, fw_b, fr_w, fr_b,
                                          bc_w, bc_b, bs_w, bs_b, outF, outB);
}
